// MCA_72473278152788
// MI455X (gfx1250) — matrix operand order repaired and checked by reading, not compiled, not run on hardware
//
#include <hip/hip_runtime.h>
#include <hip/hip_bf16.h>

typedef __attribute__((ext_vector_type(16))) _Float16 v16h;
typedef __attribute__((ext_vector_type(8)))  _Float16 v8h;
typedef __attribute__((ext_vector_type(4)))  _Float16 v4h;
typedef __attribute__((ext_vector_type(8)))  float    v8f;
typedef __attribute__((ext_vector_type(4)))  int      v4i;

#define DIMC   1024
#define HEADS  16
#define HD     64
#define TT     2048
#define BB     4
#define MTOT   (TT*BB)   /* 8192 rows */

#define GLOBAL_AS __attribute__((address_space(1)))
#define LDS_AS    __attribute__((address_space(3)))

#if defined(__has_builtin)
#if __has_builtin(__builtin_amdgcn_global_load_async_to_lds_b128)
#define USE_ASYNC_LDS 1
#endif
#endif

// 16-byte global -> LDS copy; async (ASYNCcnt-tracked) when available.
// Builtin signature (from toolchain diagnostic): (int4 AS1*, int4 AS3*, Ii, Ii)
static __device__ __forceinline__ void async_copy16(const _Float16* g,
                                                    _Float16* l) {
#ifdef USE_ASYNC_LDS
  __builtin_amdgcn_global_load_async_to_lds_b128((GLOBAL_AS v4i*)g,
                                                 (LDS_AS v4i*)l, 0, 0);
#else
  *(v8h*)l = *(const v8h*)g;
#endif
}

// Wait until at most N async copies remain outstanding (in-order completion).
template <int N> static __device__ __forceinline__ void wait_async() {
#ifdef USE_ASYNC_LDS
#if __has_builtin(__builtin_amdgcn_s_wait_asynccnt)
  __builtin_amdgcn_s_wait_asynccnt(N);
#else
  asm volatile("s_wait_asynccnt %0" ::"i"(N) : "memory");
#endif
#endif
}

static __device__ __forceinline__ v16h make_a_frag(v8h lo, v8h hi) {
  v16h a;
  #pragma unroll
  for (int i = 0; i < 8; ++i) { a[i] = lo[i]; a[8 + i] = hi[i]; }
  return a;
}

static __device__ __forceinline__ v8f wmma_f16(v16h a, v16h b, v8f c) {
  return __builtin_amdgcn_wmma_f32_16x16x32_f16(false, a, false, b, (short)0, c,
                                                false, false);
}

// ---------------------------------------------------------------------------
// fp32 (T,B,C) -> fp16 (B*T, C)  (permute + convert), 4 elems/thread
// ---------------------------------------------------------------------------
__global__ void permute_cvt_kernel(const float* __restrict__ src,
                                   _Float16* __restrict__ dst) {
  int i  = blockIdx.x * blockDim.x + threadIdx.x;       // over MTOT*DIMC/4
  int c4 = i & (DIMC / 4 - 1);
  int tb = i >> 8;                                      // DIMC/4 == 256
  int b  = tb & (BB - 1);
  int t  = tb >> 2;
  float4 f = ((const float4*)src)[i];
  v4h h = { (_Float16)f.x, (_Float16)f.y, (_Float16)f.z, (_Float16)f.w };
  ((v4h*)dst)[((long)b * TT + t) * (DIMC / 4) + c4] = h;
}

// fp32 -> fp16 straight copy (weights), 4 elems/thread
__global__ void cvt_kernel(const float* __restrict__ src,
                           _Float16* __restrict__ dst, int n4) {
  int i = blockIdx.x * blockDim.x + threadIdx.x;
  if (i < n4) {
    float4 f = ((const float4*)src)[i];
    v4h h = { (_Float16)f.x, (_Float16)f.y, (_Float16)f.z, (_Float16)f.w };
    ((v4h*)dst)[i] = h;
  }
}

// ---------------------------------------------------------------------------
// Shared GEMM mainloop:  acc(16x64) += X[m0:m0+16, :] * W[n0:n0+64, :]^T
// W tile (64x32 halves, 4 KB) staged in LDS via async copies, double-buffered
// one k-step ahead.  All 8 waves of the block share the same n0 tile.
// ---------------------------------------------------------------------------
static __device__ __forceinline__ void gemm_mainloop(
    const _Float16* __restrict__ X, const _Float16* __restrict__ W,
    _Float16* smem /* [2][64*32] */, int m0_wave, int n0_blk, v8f acc[4]) {
  const int tid   = threadIdx.x;
  const int lane  = tid & 31;
  const int lhalf = lane >> 4;
  const int l16   = lane & 15;
  const int wrow  = tid >> 2;           // 0..63  (N row of W tile)
  const int wch   = tid & 3;            // 0..3   (8-half chunk)

  const _Float16* __restrict__ xrow = X + (long)(m0_wave + l16) * DIMC;
  const _Float16* __restrict__ wsrc = W + (long)(n0_blk + wrow) * DIMC + wch * 8;
  _Float16* const my_sl = smem + wrow * 32 + wch * 8;

  async_copy16(&wsrc[0], my_sl);                 // prologue: stage k-step 0

  for (int k0 = 0; k0 < DIMC; k0 += 32) {
    const int cur = (k0 >> 5) & 1;
    if (k0 + 32 < DIMC) {
      async_copy16(&wsrc[k0 + 32], my_sl + (cur ^ 1) * (64 * 32));
      wait_async<1>();                           // all but the newest are done
    } else {
      wait_async<0>();
    }
    __syncthreads();

    // A operand: row m = m0+l16, 16 contiguous K halves (half selects K+16)
    v16h a = make_a_frag(*(const v8h*)&xrow[k0 + 16 * lhalf],
                         *(const v8h*)&xrow[k0 + 16 * lhalf + 8]);
    const _Float16* sb = smem + cur * (64 * 32);
    #pragma unroll
    for (int j = 0; j < 4; ++j) {
      // B operand: lane = column n, 16 contiguous K halves (half selects K+16)
      v16h b = *(const v16h*)&sb[(j * 16 + l16) * 32 + 16 * lhalf];
      acc[j] = wmma_f16(a, b, acc[j]);
    }
    __syncthreads();
  }
}

// ---------------------------------------------------------------------------
// Fused QKV projection.  z = 0:Q  1:K  2:V
//   Q,K stored (B,H,T,hd);  V stored transposed (B,H,hd,T)
// ---------------------------------------------------------------------------
__global__ __launch_bounds__(256) void qkv_kernel(
    const _Float16* __restrict__ X1, const _Float16* __restrict__ X2,
    const _Float16* __restrict__ Wq, const _Float16* __restrict__ Wk,
    const _Float16* __restrict__ Wv, const float* __restrict__ bq,
    const float* __restrict__ bk, const float* __restrict__ bv,
    _Float16* __restrict__ Q, _Float16* __restrict__ Kd,
    _Float16* __restrict__ Vt) {
  __shared__ __align__(32) _Float16 smem[2][64 * 32];
  const int which = blockIdx.z;
  const _Float16* X = (which == 0) ? X1 : X2;
  const _Float16* W = (which == 0) ? Wq : ((which == 1) ? Wk : Wv);
  const float* bias = (which == 0) ? bq : ((which == 1) ? bk : bv);

  const int tid = threadIdx.x, lane = tid & 31;
  const int lhalf = lane >> 4, l16 = lane & 15;
  const int n0 = blockIdx.x * 64;
  const int m0 = blockIdx.y * 128 + (tid >> 5) * 16;

  v8f acc[4];
  #pragma unroll
  for (int j = 0; j < 4; ++j)
    acc[j] = (v8f){0.f, 0.f, 0.f, 0.f, 0.f, 0.f, 0.f, 0.f};

  gemm_mainloop(X, W, &smem[0][0], m0, n0, acc);

  if (which <= 1) {
    _Float16* dst = (which == 0) ? Q : Kd;
    #pragma unroll
    for (int j = 0; j < 4; ++j) {
      int n = n0 + j * 16 + l16;
      float bv_ = bias[n];
      int h = n >> 6, d = n & (HD - 1);
      #pragma unroll
      for (int r = 0; r < 8; ++r) {
        int m = m0 + r + 8 * lhalf;
        int b = m >> 11, t = m & (TT - 1);
        dst[(((long)(b * HEADS + h) * TT + t) * HD) + d] =
            (_Float16)(acc[j][r] + bv_);
      }
    }
  } else {  // V transposed: lane's 8 accum rows are contiguous t -> b128 store
    #pragma unroll
    for (int j = 0; j < 4; ++j) {
      int n = n0 + j * 16 + l16;
      float bv_ = bias[n];
      int h = n >> 6, d = n & (HD - 1);
      int mb = m0 + 8 * lhalf;
      int b = mb >> 11, t = mb & (TT - 1);
      v8h pack;
      #pragma unroll
      for (int r = 0; r < 8; ++r) pack[r] = (_Float16)(acc[j][r] + bv_);
      *(v8h*)&Vt[((long)(b * HEADS + h) * HD + d) * TT + t] = pack;
    }
  }
}

// ---------------------------------------------------------------------------
// Flash attention.  grid = (T/128, B*H); wave handles 16 q-rows.  K/V tiles
// (64 keys) are shared by all 8 waves -> staged in LDS with async copies,
// double-buffered one tile ahead.  Computes S^T = K * Q^T so softmax rows
// live per-lane and the S^T D-fragment layout coincides with the A-operand
// layout needed for P*V (no LDS transpose of P).
// ---------------------------------------------------------------------------
__global__ __launch_bounds__(256) void attn_kernel(
    const _Float16* __restrict__ Q, const _Float16* __restrict__ Kd,
    const _Float16* __restrict__ Vt, _Float16* __restrict__ O16) {
  __shared__ __align__(32) _Float16 Ks[2][64 * 64];  // [key row][hd]
  __shared__ __align__(32) _Float16 Vs[2][64 * 64];  // [hd d][key]
  const int bh = blockIdx.y;                         // b*HEADS + h
  const int tid = threadIdx.x, lane = tid & 31;
  const int lhalf = lane >> 4, l16 = lane & 15;
  const int q0 = blockIdx.x * 128 + (tid >> 5) * 16;

  const _Float16* __restrict__ Qb = Q + (long)bh * TT * HD;
  const _Float16* __restrict__ Kb = Kd + (long)bh * TT * HD;
  const _Float16* __restrict__ Vb = Vt + (long)bh * HD * TT;

  // cooperative stage of one 64-key K/V tile (4 async b128 per thread)
  const int srow = tid >> 3;       // 0..31
  const int sch  = tid & 7;        // 0..7 (8-half chunk)
  auto stage_kv = [&](int kt, int buf) {
    #pragma unroll
    for (int i = 0; i < 2; ++i) {
      int row = srow + 32 * i;
      async_copy16(Kb + (long)(kt + row) * HD + sch * 8,
                   &Ks[buf][row * 64 + sch * 8]);
      async_copy16(Vb + (long)row * TT + kt + sch * 8,
                   &Vs[buf][row * 64 + sch * 8]);
    }
  };

  // B operand built from Q (kept in registers): lane = q column
  v16h bq_[2];
  {
    const _Float16* qrow = Qb + (long)(q0 + l16) * HD;
    #pragma unroll
    for (int s = 0; s < 2; ++s)
      bq_[s] = *(const v16h*)&qrow[s * 32 + 16 * lhalf];
  }

  v8f o[4];
  #pragma unroll
  for (int j = 0; j < 4; ++j)
    o[j] = (v8f){0.f, 0.f, 0.f, 0.f, 0.f, 0.f, 0.f, 0.f};
  float m_run = -1e30f, l_run = 0.f;

  stage_kv(0, 0);                                     // prologue

  for (int kt = 0; kt < TT; kt += 64) {
    const int cur = (kt >> 6) & 1;
    if (kt + 64 < TT) {
      stage_kv(kt + 64, cur ^ 1);
      wait_async<4>();             // everything older than the 4 new is done
    } else {
      wait_async<0>();
    }
    __syncthreads();

    // S^T tile: M = 64 key rows (4 frags), N = 16 q rows
    v8f st[4];
    #pragma unroll
    for (int mi = 0; mi < 4; ++mi)
      st[mi] = (v8f){0.f, 0.f, 0.f, 0.f, 0.f, 0.f, 0.f, 0.f};

    #pragma unroll
    for (int mi = 0; mi < 4; ++mi) {
      const _Float16* krow = &Ks[cur][(mi * 16 + l16) * 64];
      #pragma unroll
      for (int s = 0; s < 2; ++s) {
        v16h ak = make_a_frag(*(const v8h*)&krow[s * 32 + 16 * lhalf],
                              *(const v8h*)&krow[s * 32 + 16 * lhalf + 8]);
        st[mi] = wmma_f16(ak, bq_[s], st[mi]);
      }
    }

    // softmax over this key tile (row = q = l16, spread over lane^16 pair)
    float cmax = -1e30f;
    #pragma unroll
    for (int mi = 0; mi < 4; ++mi)
      #pragma unroll
      for (int r = 0; r < 8; ++r) {
        st[mi][r] *= 0.03125f;                     // 1/sqrt(1024)
        cmax = fmaxf(cmax, st[mi][r]);
      }
    cmax = fmaxf(cmax, __shfl_xor(cmax, 16, 32));

    float mnew  = fmaxf(m_run, cmax);
    float alpha = __expf(m_run - mnew);
    m_run = mnew;

    float rsum = 0.f;
    v16h ap[2];   // P as A-operand for P*V; layout matches S^T D-fragments
    #pragma unroll
    for (int s = 0; s < 2; ++s)
      #pragma unroll
      for (int e = 0; e < 8; ++e) {
        float p0 = __expf(st[2 * s][e] - mnew);
        float p1 = __expf(st[2 * s + 1][e] - mnew);
        rsum += p0 + p1;
        ap[s][e]     = (_Float16)p0;
        ap[s][8 + e] = (_Float16)p1;
      }
    rsum += __shfl_xor(rsum, 16, 32);
    l_run = l_run * alpha + rsum;

    // rescale O accumulators (row m needs alpha of q-row m, held in lane m)
    #pragma unroll
    for (int r = 0; r < 8; ++r) {
      float am = __shfl(alpha, r + 8 * lhalf, 32);
      #pragma unroll
      for (int j = 0; j < 4; ++j) o[j][r] *= am;
    }

    // O += P * V   (B operand from transposed V tile in LDS)
    #pragma unroll
    for (int s = 0; s < 2; ++s)
      #pragma unroll
      for (int j = 0; j < 4; ++j) {
        const _Float16* vrow =
            &Vs[cur][(j * 16 + l16) * 64 + s * 32 + 8 * lhalf];
        o[j] = wmma_f16(ap[s],
                        make_a_frag(*(const v8h*)vrow, *(const v8h*)(vrow + 16)),
                        o[j]);
      }
    __syncthreads();
  }

  // epilogue: scale by 1/l and store to (B,T,C) f16
  float invl = 1.0f / l_run;
  const int b = bh >> 4, h = bh & (HEADS - 1);
  #pragma unroll
  for (int r = 0; r < 8; ++r) {
    float sc = __shfl(invl, r + 8 * lhalf, 32);
    int t = q0 + r + 8 * lhalf;
    #pragma unroll
    for (int j = 0; j < 4; ++j) {
      int c = h * HD + j * 16 + l16;
      O16[((long)b * TT + t) * DIMC + c] = (_Float16)(o[j][r] * sc);
    }
  }
}

// ---------------------------------------------------------------------------
// Output projection: d_out = O16 * Wu^T + bu   (fp32 out, (B,T,C) row-major)
// ---------------------------------------------------------------------------
__global__ __launch_bounds__(256) void uproj_kernel(
    const _Float16* __restrict__ X, const _Float16* __restrict__ W,
    const float* __restrict__ bias, float* __restrict__ Y) {
  __shared__ __align__(32) _Float16 smem[2][64 * 32];
  const int tid = threadIdx.x, lane = tid & 31;
  const int lhalf = lane >> 4, l16 = lane & 15;
  const int n0 = blockIdx.x * 64;
  const int m0 = blockIdx.y * 128 + (tid >> 5) * 16;

  v8f acc[4];
  #pragma unroll
  for (int j = 0; j < 4; ++j)
    acc[j] = (v8f){0.f, 0.f, 0.f, 0.f, 0.f, 0.f, 0.f, 0.f};

  gemm_mainloop(X, W, &smem[0][0], m0, n0, acc);

  #pragma unroll
  for (int j = 0; j < 4; ++j) {
    int n = n0 + j * 16 + l16;
    float bv_ = bias[n];
    #pragma unroll
    for (int r = 0; r < 8; ++r) {
      int m = m0 + r + 8 * lhalf;
      Y[(long)m * DIMC + n] = acc[j][r] + bv_;
    }
  }
}

// ---------------------------------------------------------------------------
extern "C" void kernel_launch(void* const* d_in, const int* in_sizes, int n_in,
                              void* d_out, int out_size, void* d_ws,
                              size_t ws_size, hipStream_t stream) {
  const float* x1 = (const float*)d_in[0];
  const float* x2 = (const float*)d_in[1];
  const float* Wq = (const float*)d_in[2];
  const float* bq = (const float*)d_in[3];
  const float* Wk = (const float*)d_in[4];
  const float* bk = (const float*)d_in[5];
  const float* Wv = (const float*)d_in[6];
  const float* bv = (const float*)d_in[7];
  const float* Wu = (const float*)d_in[8];
  const float* bu = (const float*)d_in[9];
  float* out = (float*)d_out;

  char* ws = (char*)d_ws;
  size_t off = 0;
  auto take = [&](size_t bytes) {
    void* p = ws + off;
    off += (bytes + 255) & ~(size_t)255;
    return p;
  };
  const size_t act_b = (size_t)MTOT * DIMC * sizeof(_Float16);  // 16 MB
  const size_t w_b   = (size_t)DIMC * DIMC * sizeof(_Float16);  //  2 MB
  _Float16* X1  = (_Float16*)take(act_b);
  _Float16* X2  = (_Float16*)take(act_b);
  _Float16* Qh  = (_Float16*)take(act_b);
  _Float16* Kh  = (_Float16*)take(act_b);
  _Float16* Vth = (_Float16*)take(act_b);
  _Float16* Oh  = (_Float16*)take(act_b);
  _Float16* W16q = (_Float16*)take(w_b);
  _Float16* W16k = (_Float16*)take(w_b);
  _Float16* W16v = (_Float16*)take(w_b);
  _Float16* W16u = (_Float16*)take(w_b);
  (void)ws_size;

  const int n4w = DIMC * DIMC / 4;
  permute_cvt_kernel<<<MTOT * DIMC / 4 / 256, 256, 0, stream>>>(x1, X1);
  permute_cvt_kernel<<<MTOT * DIMC / 4 / 256, 256, 0, stream>>>(x2, X2);
  cvt_kernel<<<n4w / 256, 256, 0, stream>>>(Wq, W16q, n4w);
  cvt_kernel<<<n4w / 256, 256, 0, stream>>>(Wk, W16k, n4w);
  cvt_kernel<<<n4w / 256, 256, 0, stream>>>(Wv, W16v, n4w);
  cvt_kernel<<<n4w / 256, 256, 0, stream>>>(Wu, W16u, n4w);

  qkv_kernel<<<dim3(DIMC / 64, MTOT / 128, 3), 256, 0, stream>>>(
      X1, X2, W16q, W16k, W16v, bq, bk, bv, Qh, Kh, Vth);

  attn_kernel<<<dim3(TT / 128, BB * HEADS), 256, 0, stream>>>(Qh, Kh, Vth, Oh);

  uproj_kernel<<<dim3(DIMC / 64, MTOT / 128), 256, 0, stream>>>(Oh, W16u, bu,
                                                                out);
}
